// ContextQueryAttention_30245159698596
// MI455X (gfx1250) — matrix operand order repaired and checked by reading, not compiled, not run on hardware
//
#include <hip/hip_runtime.h>
#include <hip/hip_bf16.h>

// ContextQueryAttention for MI455X (gfx1250), wave32 + WMMA bf16 16x16x32
// + async global->LDS (ASYNCcnt) staging for copy-only GEMM operands.
//
// Rewrite: (S1 @ S2^T) @ Ct == S1 @ (S2^T @ Ct)  -> 17 GFLOP instead of 60.
// GEMMs: bf16 operands, fp32 accumulation. Softmax/exp/bias/epilogues in fp32.

typedef __attribute__((ext_vector_type(16))) __bf16 v16bf;
typedef __attribute__((ext_vector_type(8)))  float  v8f;

constexpr int NB     = 64;
constexpr int DMODEL = 128;
constexpr int LCC    = 1024;
constexpr int LQQ    = 256;
constexpr float NEGINF = -1.0e30f;
constexpr int KLD = 40;   // 32 K values + 8 bf16 pad per staged row (80B stride)

__device__ inline v8f wmma_bf(v16bf a, v16bf b, v8f c) {
  return __builtin_amdgcn_wmma_f32_16x16x32_bf16(false, a, false, b, (short)0, c,
                                                 false, false);
}

// A staged as [M][KLD]; lane needs K = {kb..kb+7, kb+16..kb+23}, kb = 8*(lane>=16)
__device__ inline v16bf frag_a(const __bf16* A, int lane, int m) {
  const __bf16* p = A + (m + (lane & 15)) * KLD + ((lane & 16) >> 1);
  v16bf r;
#pragma unroll
  for (int i = 0; i < 8; ++i) r[i] = p[i];
#pragma unroll
  for (int i = 8; i < 16; ++i) r[i] = p[8 + i];
  return r;
}

// B staged as [N][KLD]; lane needs K = {kb..kb+7, kb+16..kb+23}, kb = 8*(lane>=16)
__device__ inline v16bf frag_b(const __bf16* B, int lane, int n) {
  const __bf16* p = B + (n + (lane & 15)) * KLD + ((lane & 16) >> 1);
  v16bf r;
#pragma unroll
  for (int i = 0; i < 8; ++i) r[i] = p[i];
#pragma unroll
  for (int i = 8; i < 16; ++i) r[i] = p[8 + i];
  return r;
}

// 16B per-lane async copy: LDS[lds_off..+15] = MEM[src..+15]  (ASYNCcnt)
__device__ inline void async_b128(unsigned lds_off, const __bf16* src) {
  asm volatile("global_load_async_to_lds_b128 %0, %1, off"
               :: "v"(lds_off), "v"(src) : "memory");
}
__device__ inline void wait_async0() {
  asm volatile("s_wait_asynccnt 0x0" ::: "memory");
}
__device__ inline unsigned lds_addr(const void* p) {
  return (unsigned)(size_t)p;   // low 32 bits of generic shared ptr = LDS offset
}

// ---------------------------------------------------------------- bias vectors
__global__ __launch_bounds__(256) void k_bias(const float* __restrict__ C,
                                              const float* __restrict__ Q,
                                              const float* __restrict__ w,
                                              float* __restrict__ cbias,
                                              float* __restrict__ qbias) {
  int gid = blockIdx.x * 256 + threadIdx.x;
  if (gid < NB * LCC) {
    int b = gid >> 10, l = gid & (LCC - 1);
    const float* Cb = C + (size_t)b * DMODEL * LCC;
    float s = 0.f;
#pragma unroll 4
    for (int d = 0; d < DMODEL; ++d) s += Cb[d * LCC + l] * w[d];
    cbias[gid] = s;
  } else {
    int idx = gid - NB * LCC;
    if (idx < NB * LQQ) {
      int b = idx >> 8, m = idx & (LQQ - 1);
      const float* Qb = Q + (size_t)b * DMODEL * LQQ;
      float s = 0.f;
#pragma unroll 4
      for (int d = 0; d < DMODEL; ++d) s += Qb[d * LQQ + m] * w[DMODEL + d];
      qbias[idx] = s;
    }
  }
}

// --------------------------------------- fp32 -> bf16 copies of C and Q (once)
__global__ __launch_bounds__(256) void k_convert(const float* __restrict__ C,
                                                 const float* __restrict__ Q,
                                                 __bf16* __restrict__ Cbf,
                                                 __bf16* __restrict__ Qbf) {
  const int NC = NB * DMODEL * LCC;
  const int NQ = NB * DMODEL * LQQ;
  int idx = blockIdx.x * 256 + threadIdx.x;
  int stride = gridDim.x * 256;
  for (int i = idx; i < NC + NQ; i += stride) {
    if (i < NC) Cbf[i] = (__bf16)C[i];
    else        Qbf[i - NC] = (__bf16)Q[i - NC];
  }
}

// -------------------------------------------- S = cbias + qbias + (Ct*w3)@Qt^T
__global__ __launch_bounds__(256) void k_scores(const float* __restrict__ C,
                                                const float* __restrict__ Q,
                                                const float* __restrict__ w,
                                                const float* __restrict__ cbias,
                                                const float* __restrict__ qbias,
                                                float* __restrict__ S) {
  __shared__ union {
    struct { __bf16 a[128 * KLD]; __bf16 b[64 * KLD]; } st;
    float o[128 * 65];
  } sm;
  const int b  = blockIdx.z;
  const int l0 = blockIdx.x * 128;
  const int m0 = blockIdx.y * 64;
  const int t = threadIdx.x, lane = t & 31, wv = t >> 5;
  const int wm = wv & 3;    // 4 waves over M=128
  const int wn = wv >> 2;   // 2 waves over N=64
  const float* Cb = C + (size_t)b * DMODEL * LCC;
  const float* Qb = Q + (size_t)b * DMODEL * LQQ;
  const float* w3 = w + 2 * DMODEL;
  v8f acc[2][2];
#pragma unroll
  for (int mi = 0; mi < 2; ++mi)
#pragma unroll
    for (int ni = 0; ni < 2; ++ni)
#pragma unroll
      for (int j = 0; j < 8; ++j) acc[mi][ni][j] = 0.f;

  for (int kk = 0; kk < DMODEL; kk += 32) {
    __syncthreads();
#pragma unroll
    for (int r = 0; r < 16; ++r) {              // A: (Ct*w3) 32k x 128m -> [m][k]
      int idx = t + 256 * r;
      int k = idx >> 7, ml = idx & 127;
      sm.st.a[ml * KLD + k] = (__bf16)(Cb[(kk + k) * LCC + l0 + ml] * w3[kk + k]);
    }
#pragma unroll
    for (int r = 0; r < 8; ++r) {               // B: Qt^T 32k x 64n -> [n][k]
      int idx = t + 256 * r;
      int k = idx >> 6, nl = idx & 63;
      sm.st.b[nl * KLD + k] = (__bf16)(Qb[(kk + k) * LQQ + m0 + nl]);
    }
    __syncthreads();
    v16bf af0 = frag_a(sm.st.a, lane, wm * 32 + 0);
    v16bf af1 = frag_a(sm.st.a, lane, wm * 32 + 16);
    v16bf bf0 = frag_b(sm.st.b, lane, wn * 32 + 0);
    v16bf bf1 = frag_b(sm.st.b, lane, wn * 32 + 16);
    acc[0][0] = wmma_bf(af0, bf0, acc[0][0]);
    acc[0][1] = wmma_bf(af0, bf1, acc[0][1]);
    acc[1][0] = wmma_bf(af1, bf0, acc[1][0]);
    acc[1][1] = wmma_bf(af1, bf1, acc[1][1]);
  }
  __syncthreads();
  const int hi8 = (lane & 16) >> 1, nc = lane & 15;
#pragma unroll
  for (int mi = 0; mi < 2; ++mi)
#pragma unroll
    for (int ni = 0; ni < 2; ++ni)
#pragma unroll
      for (int j = 0; j < 8; ++j) {
        int row = wm * 32 + mi * 16 + j + hi8;
        int col = wn * 32 + ni * 16 + nc;
        sm.o[row * 65 + col] = acc[mi][ni][j];
      }
  __syncthreads();
#pragma unroll 4
  for (int r = 0; r < 32; ++r) {                // coalesced (m-contiguous) stores
    int l = r * 4 + (t >> 6);
    int m = t & 63;
    float v = sm.o[l * 65 + m] + cbias[b * LCC + l0 + l] + qbias[b * LQQ + m0 + m];
    S[((size_t)b * LCC + l0 + l) * LQQ + m0 + m] = v;
  }
}

// ---------------------------------------------------- softmax stats over m (S1)
__global__ __launch_bounds__(256) void k_rowstats(const float* __restrict__ S,
                                                  const float* __restrict__ qmask,
                                                  float* __restrict__ rowmax,
                                                  float* __restrict__ rowinv) {
  int wv = threadIdx.x >> 5, lane = threadIdx.x & 31;
  int row = blockIdx.x * 8 + wv;                // [0, NB*LCC)
  int b = row >> 10;
  const float* Sr = S + (size_t)row * LQQ;
  const float* qm = qmask + b * LQQ;
  float x[8], mx = -3.0e38f;
#pragma unroll
  for (int i = 0; i < 8; ++i) {
    int m = lane + 32 * i;
    float v = Sr[m] + (1.0f - qm[m]) * NEGINF;
    x[i] = v;
    mx = fmaxf(mx, v);
  }
#pragma unroll
  for (int off = 16; off > 0; off >>= 1) mx = fmaxf(mx, __shfl_xor(mx, off, 32));
  float s = 0.f;
#pragma unroll
  for (int i = 0; i < 8; ++i) s += __expf(x[i] - mx);
#pragma unroll
  for (int off = 16; off > 0; off >>= 1) s += __shfl_xor(s, off, 32);
  if (lane == 0) { rowmax[row] = mx; rowinv[row] = 1.0f / s; }
}

// ---------------------------------------------------- softmax stats over l (S2)
__global__ __launch_bounds__(256) void k_colstats(const float* __restrict__ S,
                                                  const float* __restrict__ cmask,
                                                  float* __restrict__ colmax,
                                                  float* __restrict__ colinv) {
  int idx = blockIdx.x * 256 + threadIdx.x;     // [0, NB*LQQ)
  int b = idx >> 8, m = idx & (LQQ - 1);
  const float* Sb = S + (size_t)b * LCC * LQQ;
  const float* cm = cmask + b * LCC;
  float mx = -3.0e38f, s = 0.f;
  for (int l = 0; l < LCC; ++l) {               // online max/sum, coalesced in m
    float v = Sb[(size_t)l * LQQ + m] + (1.0f - cm[l]) * NEGINF;
    float nm = fmaxf(mx, v);
    s = s * __expf(mx - nm) + __expf(v - nm);
    mx = nm;
  }
  colmax[idx] = mx;
  colinv[idx] = 1.0f / s;
}

// --------------------------- T' = (S2^T @ Ct) stored bf16 transposed as [d][m]
__global__ __launch_bounds__(256) void k_tmat(const float* __restrict__ S,
                                              const __bf16* __restrict__ Cbf,
                                              const float* __restrict__ cmask,
                                              const float* __restrict__ colmax,
                                              const float* __restrict__ colinv,
                                              __bf16* __restrict__ Tbf) {
  __shared__ union {
    struct { __bf16 a[64 * KLD]; __bf16 b[128 * KLD]; } st;
    __bf16 o[128 * 72];                         // [d][m] transpose stage
  } sm;
  const int b  = blockIdx.z;
  const int m0 = blockIdx.x * 64;
  const int t = threadIdx.x, lane = t & 31, wv = t >> 5;
  const int wm = wv & 1;    // 2 waves over M=64 (m)
  const int wn = wv >> 1;   // 4 waves over N=128 (d)
  const float*  Sb  = S + (size_t)b * LCC * LQQ;
  const __bf16* Cb  = Cbf + (size_t)b * DMODEL * LCC;
  const float* cmx = colmax + b * LQQ + m0;
  const float* cin = colinv + b * LQQ + m0;
  const float* cmk = cmask + b * LCC;
  v8f acc[2][2];
#pragma unroll
  for (int mi = 0; mi < 2; ++mi)
#pragma unroll
    for (int ni = 0; ni < 2; ++ni)
#pragma unroll
      for (int j = 0; j < 8; ++j) acc[mi][ni][j] = 0.f;

  for (int kk = 0; kk < LCC; kk += 32) {
    __syncthreads();
#pragma unroll
    for (int r = 0; r < 2; ++r) {               // B: Ct tile, async 16B copies
      int idx = t + 256 * r;                    // 512 chunks of 8 k-values
      int n = idx >> 2, kc = (idx & 3) << 3;
      async_b128(lds_addr(&sm.st.b[n * KLD + kc]), Cb + (size_t)n * LCC + kk + kc);
    }
#pragma unroll
    for (int r = 0; r < 8; ++r) {               // A: exp(S - colmax)*colinv
      int idx = t + 256 * r;
      int k = idx >> 6, ml = idx & 63;
      int l = kk + k;
      float v = Sb[(size_t)l * LQQ + m0 + ml] + (1.0f - cmk[l]) * NEGINF;
      sm.st.a[ml * KLD + k] = (__bf16)(__expf(v - cmx[ml]) * cin[ml]);
    }
    wait_async0();
    __syncthreads();
    v16bf af0 = frag_a(sm.st.a, lane, wm * 32 + 0);
    v16bf af1 = frag_a(sm.st.a, lane, wm * 32 + 16);
    v16bf bf0 = frag_b(sm.st.b, lane, wn * 32 + 0);
    v16bf bf1 = frag_b(sm.st.b, lane, wn * 32 + 16);
    acc[0][0] = wmma_bf(af0, bf0, acc[0][0]);
    acc[0][1] = wmma_bf(af0, bf1, acc[0][1]);
    acc[1][0] = wmma_bf(af1, bf0, acc[1][0]);
    acc[1][1] = wmma_bf(af1, bf1, acc[1][1]);
  }
  __syncthreads();
  const int hi8 = (lane & 16) >> 1, nc = lane & 15;
#pragma unroll
  for (int mi = 0; mi < 2; ++mi)
#pragma unroll
    for (int ni = 0; ni < 2; ++ni)
#pragma unroll
      for (int j = 0; j < 8; ++j) {
        int mrow = wm * 32 + mi * 16 + j + hi8;
        int d    = wn * 32 + ni * 16 + nc;
        sm.o[d * 72 + mrow] = (__bf16)acc[mi][ni][j];
      }
  __syncthreads();
  __bf16* Tb = Tbf + (size_t)b * DMODEL * LQQ;
#pragma unroll 4
  for (int r = 0; r < 32; ++r) {                // coalesced m-contiguous stores
    int d = r * 4 + (t >> 6);
    int m = t & 63;
    Tb[(size_t)d * LQQ + m0 + m] = sm.o[d * 72 + m];
  }
}

// --------------- fused A = S1@Qt, Bv = S1@T, + concat/transpose output epilogue
__global__ __launch_bounds__(256) void k_out(const float* __restrict__ S,
                                             const float* __restrict__ C,
                                             const __bf16* __restrict__ Qbf,
                                             const __bf16* __restrict__ Tbf,
                                             const float* __restrict__ qmask,
                                             const float* __restrict__ rowmax,
                                             const float* __restrict__ rowinv,
                                             float* __restrict__ out) {
  __shared__ union {
    struct { __bf16 a[64 * KLD]; __bf16 bq[128 * KLD]; __bf16 bt[128 * KLD]; } st;
    struct { float a[128 * 65]; float v[128 * 65]; } o;   // [d][l] transposed
  } sm;
  const int b  = blockIdx.z;
  const int l0 = blockIdx.x * 64;
  const int t = threadIdx.x, lane = t & 31, wv = t >> 5;
  const int wm = wv & 1;    // 2 waves over M=64 (l)
  const int wn = wv >> 1;   // 4 waves over N=128 (d)
  const float*  Sb  = S + (size_t)b * LCC * LQQ;
  const float*  Cb  = C + (size_t)b * DMODEL * LCC;
  const __bf16* Qb  = Qbf + (size_t)b * DMODEL * LQQ;
  const __bf16* Tb  = Tbf + (size_t)b * DMODEL * LQQ;
  const float* rmx = rowmax + b * LCC + l0;
  const float* rin = rowinv + b * LCC + l0;
  const float* qmk = qmask + b * LQQ;
  v8f accA[2][2], accV[2][2];
#pragma unroll
  for (int mi = 0; mi < 2; ++mi)
#pragma unroll
    for (int ni = 0; ni < 2; ++ni)
#pragma unroll
      for (int j = 0; j < 8; ++j) { accA[mi][ni][j] = 0.f; accV[mi][ni][j] = 0.f; }

  for (int kk = 0; kk < LQQ; kk += 32) {
    __syncthreads();
#pragma unroll
    for (int r = 0; r < 2; ++r) {               // BQ: Qt tile, async 16B copies
      int idx = t + 256 * r;
      int n = idx >> 2, kc = (idx & 3) << 3;
      async_b128(lds_addr(&sm.st.bq[n * KLD + kc]), Qb + (size_t)n * LQQ + kk + kc);
    }
#pragma unroll
    for (int r = 0; r < 2; ++r) {               // BT: T' tile, async 16B copies
      int idx = t + 256 * r;
      int n = idx >> 2, kc = (idx & 3) << 3;
      async_b128(lds_addr(&sm.st.bt[n * KLD + kc]), Tb + (size_t)n * LQQ + kk + kc);
    }
#pragma unroll
    for (int r = 0; r < 8; ++r) {               // A: S1 = exp(S - rowmax)*rowinv
      int idx = t + 256 * r;
      int ml = idx >> 5, k = idx & 31;
      int m = kk + k;
      float v = Sb[(size_t)(l0 + ml) * LQQ + m] + (1.0f - qmk[m]) * NEGINF;
      sm.st.a[ml * KLD + k] = (__bf16)(__expf(v - rmx[ml]) * rin[ml]);
    }
    wait_async0();
    __syncthreads();
    v16bf af0 = frag_a(sm.st.a, lane, wm * 32 + 0);
    v16bf af1 = frag_a(sm.st.a, lane, wm * 32 + 16);
    v16bf q0  = frag_b(sm.st.bq, lane, wn * 32 + 0);
    v16bf q1  = frag_b(sm.st.bq, lane, wn * 32 + 16);
    v16bf t0  = frag_b(sm.st.bt, lane, wn * 32 + 0);
    v16bf t1  = frag_b(sm.st.bt, lane, wn * 32 + 16);
    accA[0][0] = wmma_bf(af0, q0, accA[0][0]);
    accA[0][1] = wmma_bf(af0, q1, accA[0][1]);
    accA[1][0] = wmma_bf(af1, q0, accA[1][0]);
    accA[1][1] = wmma_bf(af1, q1, accA[1][1]);
    accV[0][0] = wmma_bf(af0, t0, accV[0][0]);
    accV[0][1] = wmma_bf(af0, t1, accV[0][1]);
    accV[1][0] = wmma_bf(af1, t0, accV[1][0]);
    accV[1][1] = wmma_bf(af1, t1, accV[1][1]);
  }
  __syncthreads();
  const int hi8 = (lane & 16) >> 1, nc = lane & 15;
#pragma unroll
  for (int mi = 0; mi < 2; ++mi)
#pragma unroll
    for (int ni = 0; ni < 2; ++ni)
#pragma unroll
      for (int j = 0; j < 8; ++j) {
        int lrow = wm * 32 + mi * 16 + j + hi8;
        int d    = wn * 32 + ni * 16 + nc;
        sm.o.a[d * 65 + lrow] = accA[mi][ni][j];
        sm.o.v[d * 65 + lrow] = accV[mi][ni][j];
      }
  __syncthreads();
  float* ob = out + (size_t)b * 4 * DMODEL * LCC;
#pragma unroll 2
  for (int r = 0; r < 32; ++r) {                // l-contiguous coalesced stores
    int d = r * 4 + (t >> 6);
    int l = t & 63;
    float cv = Cb[(size_t)d * LCC + l0 + l];
    float av = sm.o.a[d * 65 + l];
    float vv = sm.o.v[d * 65 + l];
    size_t col = (size_t)l0 + l;
    ob[(size_t)d * LCC + col]                = cv;        // Ct
    ob[((size_t)DMODEL + d) * LCC + col]     = av;        // A
    ob[((size_t)2 * DMODEL + d) * LCC + col] = cv * av;   // Ct*A
    ob[((size_t)3 * DMODEL + d) * LCC + col] = cv * vv;   // Ct*Bv
  }
}

extern "C" void kernel_launch(void* const* d_in, const int* in_sizes, int n_in,
                              void* d_out, int out_size, void* d_ws, size_t ws_size,
                              hipStream_t stream) {
  (void)in_sizes; (void)n_in; (void)out_size; (void)ws_size;
  const float* C     = (const float*)d_in[0];
  const float* Q     = (const float*)d_in[1];
  const float* cmask = (const float*)d_in[2];
  const float* qmask = (const float*)d_in[3];
  const float* w     = (const float*)d_in[4];
  float* out = (float*)d_out;

  float* ws = (float*)d_ws;                       // ~93 MB used
  float* S      = ws;                             // NB*LCC*LQQ f32
  float* rowmax = S + (size_t)NB * LCC * LQQ;     // NB*LCC
  float* rowinv = rowmax + (size_t)NB * LCC;
  float* colmax = rowinv + (size_t)NB * LCC;      // NB*LQQ
  float* colinv = colmax + (size_t)NB * LQQ;
  float* cbias  = colinv + (size_t)NB * LQQ;      // NB*LCC
  float* qbias  = cbias + (size_t)NB * LCC;       // NB*LQQ
  __bf16* Cbf   = (__bf16*)(qbias + (size_t)NB * LQQ);   // NB*DMODEL*LCC bf16
  __bf16* Qbf   = Cbf + (size_t)NB * DMODEL * LCC;       // NB*DMODEL*LQQ bf16
  __bf16* Tbf   = Qbf + (size_t)NB * DMODEL * LQQ;       // NB*DMODEL*LQQ bf16

  k_bias<<<(NB * LCC + NB * LQQ) / 256, 256, 0, stream>>>(C, Q, w, cbias, qbias);
  k_convert<<<5120, 256, 0, stream>>>(C, Q, Cbf, Qbf);
  k_scores<<<dim3(LCC / 128, LQQ / 64, NB), 256, 0, stream>>>(C, Q, w, cbias, qbias, S);
  k_rowstats<<<NB * LCC / 8, 256, 0, stream>>>(S, qmask, rowmax, rowinv);
  k_colstats<<<NB * LQQ / 256, 256, 0, stream>>>(S, cmask, colmax, colinv);
  k_tmat<<<dim3(LQQ / 64, 1, NB), 256, 0, stream>>>(S, Cbf, cmask, colmax, colinv, Tbf);
  k_out<<<dim3(LCC / 64, 1, NB), 256, 0, stream>>>(S, C, Qbf, Tbf, qmask, rowmax,
                                                   rowinv, out);
}
